// RegularAttention_23914377904900
// MI455X (gfx1250) — matrix operand order repaired and checked by reading, not compiled, not run on hardware
//
#include <hip/hip_runtime.h>

// Sliding-window (causal, width 256) attention for B=4,H=16,S=4096,D=64, fp32 I/O.
// Flash-attention style, bf16 WMMA (v_wmma_f32_16x16x32_bf16) on gfx1250 wave32.
// Double-buffered LDS staging; DPP16 butterfly reductions; scalar-uniform
// mask-free fast path; single workgroup barrier per iteration.

#define S_LEN  4096
#define D_HEAD 64
#define WIN    256
#define KPAD   72   // Ksh row stride (ushorts): 144B, 16B aligned, conflict-free
#define VPAD   40   // VshT row stride: 80B
#define PPAD   40   // Psh row stride: 80B

typedef __attribute__((ext_vector_type(16))) __bf16 v16bf;
typedef __attribute__((ext_vector_type(8)))  float  v8f;

union AFrag {
  v16bf          v;
  unsigned short u[16];
  uint4          q4[2];
};

__device__ __forceinline__ unsigned short f2bf(float f) {
  __bf16 h = (__bf16)f;                       // v_cvt_*_bf16_f32, RNE
  return __builtin_bit_cast(unsigned short, h);
}

// Butterfly reduction over each 16-lane half using DPP16 row controls.
// XOR masks {15,7,3,1} == row_mirror, row_half_mirror, quad_perm[3,2,1,0],
// quad_perm[1,0,3,2]; spans all 16 lanes of the half.
#define DPP_STEP(x, ctrl, op)                                                  \
  do {                                                                         \
    int _s = __builtin_amdgcn_update_dpp(0, __float_as_int(x), (ctrl), 0xf,    \
                                         0xf, true);                           \
    (x) = op((x), __int_as_float(_s));                                         \
  } while (0)

__device__ __forceinline__ float addf(float a, float b) { return a + b; }

__device__ __forceinline__ float dpp_max16(float x) {
  DPP_STEP(x, 0x140, fmaxf);   // row_mirror        : xor 15
  DPP_STEP(x, 0x141, fmaxf);   // row_half_mirror   : xor 7
  DPP_STEP(x, 0x01B, fmaxf);   // quad_perm[3,2,1,0]: xor 3
  DPP_STEP(x, 0x0B1, fmaxf);   // quad_perm[1,0,3,2]: xor 1
  return x;
}

__device__ __forceinline__ float dpp_sum16(float x) {
  DPP_STEP(x, 0x140, addf);
  DPP_STEP(x, 0x141, addf);
  DPP_STEP(x, 0x01B, addf);
  DPP_STEP(x, 0x0B1, addf);
  return x;
}

struct KVRegs { float4 k[4]; float4 v[4]; };

// GUARD=true only for chunks that can touch key < 0 (first blocks); uniform.
template <bool GUARD>
__device__ __forceinline__ void kv_load(const float* __restrict__ Kg,
                                        const float* __restrict__ Vg,
                                        size_t bhOff, int kc, int tid,
                                        KVRegs& r) {
  #pragma unroll
  for (int it = 0; it < 4; ++it) {
    const int lin  = tid + 128 * it;      // 0..511
    const int keyL = lin >> 4;            // 0..31
    const int d0   = (lin & 15) * 4;      // 0,4,..,60
    const int key  = kc + keyL;
    float4 kv = {0.f, 0.f, 0.f, 0.f};
    float4 vv = {0.f, 0.f, 0.f, 0.f};
    if (!GUARD || key >= 0) {
      kv = *(const float4*)(Kg + bhOff + (size_t)key * D_HEAD + d0);
      vv = *(const float4*)(Vg + bhOff + (size_t)key * D_HEAD + d0);
    }
    r.k[it] = kv;
    r.v[it] = vv;
  }
}

__device__ __forceinline__ void kv_store(unsigned short (*Ksh)[KPAD],
                                         unsigned short (*VshT)[VPAD],
                                         int tid, const KVRegs& r) {
  #pragma unroll
  for (int it = 0; it < 4; ++it) {
    const int lin  = tid + 128 * it;
    const int keyL = lin >> 4;
    const int d0   = (lin & 15) * 4;
    uint2 pk;
    pk.x = (unsigned)f2bf(r.k[it].x) | ((unsigned)f2bf(r.k[it].y) << 16);
    pk.y = (unsigned)f2bf(r.k[it].z) | ((unsigned)f2bf(r.k[it].w) << 16);
    *(uint2*)&Ksh[keyL][d0] = pk;
    VshT[d0 + 0][keyL] = f2bf(r.v[it].x);
    VshT[d0 + 1][keyL] = f2bf(r.v[it].y);
    VshT[d0 + 2][keyL] = f2bf(r.v[it].z);
    VshT[d0 + 3][keyL] = f2bf(r.v[it].w);
  }
}

__launch_bounds__(128, 2)
__global__ void swa_flash_wmma_kernel(const float* __restrict__ Qg,
                                      const float* __restrict__ Kg,
                                      const float* __restrict__ Vg,
                                      float* __restrict__ Og) {
  __shared__ unsigned short Ksh[2][32][KPAD];       // [buf][key][d]  bf16
  __shared__ unsigned short VshT[2][D_HEAD][VPAD];  // [buf][d][key]  bf16 (T)
  __shared__ unsigned short Psh[4][16][PPAD];       // per-wave P     bf16

  const int tid  = threadIdx.x;
  const int w    = tid >> 5;        // wave id 0..3
  const int lane = tid & 31;
  const int half = lane >> 4;       // 0: lanes 0-15, 1: lanes 16-31
  const int n16  = lane & 15;

  const int    qbase = blockIdx.x * 64;
  const size_t bhOff = (size_t)blockIdx.y * (size_t)S_LEN * D_HEAD;

  // ---- Load Q tile (16 rows per wave) directly into A-fragments (bf16) ----
  const float* qrow = Qg + bhOff + (size_t)(qbase + 16 * w + n16) * D_HEAD;
  AFrag qa[2];
  #pragma unroll
  for (int f = 0; f < 2; ++f) {
    const int g0 = 32 * f + 8 * half;
    #pragma unroll
    for (int e = 0; e < 8; ++e) {
      qa[f].u[e]     = f2bf(qrow[g0 + e]);        // K = g0 .. g0+7
      qa[f].u[8 + e] = f2bf(qrow[g0 + 16 + e]);   // K = g0+16 .. g0+23
    }
  }

  v8f acc[4];
  float mrow[8], lrow[8];
  #pragma unroll
  for (int t = 0; t < 4; ++t)
    #pragma unroll
    for (int e = 0; e < 8; ++e) acc[t][e] = 0.0f;
  #pragma unroll
  for (int r = 0; r < 8; ++r) { mrow[r] = -1e30f; lrow[r] = 0.0f; }

  const int q0 = qbase + 16 * w;     // first query row of this wave
  const int c0 = (qbase >> 5) - 8;   // first 32-key chunk index (may be < 0)

  // ---- prologue: stage chunk 0 into buffer 0 ----
  KVRegs pre;
  if (c0 >= 0) kv_load<false>(Kg, Vg, bhOff, c0 * 32, tid, pre);
  else         kv_load<true >(Kg, Vg, bhOff, c0 * 32, tid, pre);
  kv_store(Ksh[0], VshT[0], tid, pre);
  __syncthreads();

  for (int t = 0; t < 10; ++t) {
    const int kc  = (c0 + t) * 32;   // absolute key index of chunk start
    const int buf = t & 1;

    // Prefetch next chunk's fp32 data into registers (latency hidden
    // behind this iteration's WMMA + softmax work). kc is uniform, so the
    // key>=0 guard is a scalar branch; common path is unguarded b128 loads.
    if (t < 9) {
      if (kc + 32 >= 0) kv_load<false>(Kg, Vg, bhOff, kc + 32, tid, pre);
      else              kv_load<true >(Kg, Vg, bhOff, kc + 32, tid, pre);
    }

    // ---- scores = Q @ K^T : two 16x16 tiles (32 keys), contraction D=64 ----
    v8f s[2];
    #pragma unroll
    for (int tl = 0; tl < 2; ++tl) {
      #pragma unroll
      for (int e = 0; e < 8; ++e) s[tl][e] = 0.0f;
      #pragma unroll
      for (int dh = 0; dh < 2; ++dh) {
        AFrag bk;  // B: column n = key (kc+16*tl+n16), k = d (32*dh + {8*half+0..7, 16+8*half+0..7})
        const unsigned short* src = &Ksh[buf][16 * tl + n16][32 * dh + 8 * half];
        bk.q4[0] = *(const uint4*)(src);
        bk.q4[1] = *(const uint4*)(src + 16);
        s[tl] = __builtin_amdgcn_wmma_f32_16x16x32_bf16(
            false, qa[dh].v, false, bk.v, (short)0, s[tl], false, false);
      }
    }

    // ---- online softmax; band mask only on boundary chunks --------------
    // Chunk fully inside the band iff all (i,j) satisfy 0 <= i-j < 256:
    //   kc >= 0 && kc >= q0-240 && kc+31 <= q0.
    // Wave-uniform by construction; force scalar so this is an s_cbranch
    // (no EXEC manipulation, no phi-mov chains).
    const bool fullband = __builtin_amdgcn_readfirstlane(
        (int)((kc >= 0) & (kc >= q0 - 240) & (kc <= q0 - 31))) != 0;

    if (fullband) {
      // -------- fast path: no mask arithmetic (7 of 10 chunks) --------
      #pragma unroll
      for (int r = 0; r < 8; ++r) {
        const float s0 = s[0][r];
        const float s1 = s[1][r];
        const float rmax  = dpp_max16(fmaxf(s0, s1));
        const float mo    = mrow[r];
        const float mn    = fmaxf(mo, rmax);
        const float scale = __expf(mo - mn);
        mrow[r] = mn;
        const float p0 = __expf(s0 - mn);
        const float p1 = __expf(s1 - mn);
        lrow[r] = lrow[r] * scale + dpp_sum16(p0 + p1);
        acc[0][r] *= scale; acc[1][r] *= scale;
        acc[2][r] *= scale; acc[3][r] *= scale;
        Psh[w][8 * half + r][n16]      = f2bf(p0);
        Psh[w][8 * half + r][16 + n16] = f2bf(p1);
      }
    } else {
      // -------- masked path: boundary chunks --------
      const int  ibase = q0 + 8 * half;
      const int  j0    = kc + n16;
      const int  j1    = kc + 16 + n16;
      const bool jv0   = (j0 >= 0);           // r-independent, hoisted
      const bool jv1   = (j1 >= 0);
      #pragma unroll
      for (int r = 0; r < 8; ++r) {
        const int i = ibase + r;
        const bool ok0 = jv0 && ((unsigned)(i - j0) < (unsigned)WIN);
        const bool ok1 = jv1 && ((unsigned)(i - j1) < (unsigned)WIN);
        const float s0 = ok0 ? s[0][r] : -1e30f;
        const float s1 = ok1 ? s[1][r] : -1e30f;
        const float rmax  = dpp_max16(fmaxf(s0, s1));
        const float mo    = mrow[r];
        const float mn    = fmaxf(mo, rmax);
        const float scale = __expf(mo - mn);
        mrow[r] = mn;
        const float p0 = ok0 ? __expf(s0 - mn) : 0.0f;
        const float p1 = ok1 ? __expf(s1 - mn) : 0.0f;
        lrow[r] = lrow[r] * scale + dpp_sum16(p0 + p1);
        acc[0][r] *= scale; acc[1][r] *= scale;
        acc[2][r] *= scale; acc[3][r] *= scale;
        Psh[w][8 * half + r][n16]      = f2bf(p0);
        Psh[w][8 * half + r][16 + n16] = f2bf(p1);
      }
    }

    // Psh is per-wave: same-wave DS ops are processed in-order (ISA 7.3),
    // so only compiler ordering is needed here, not a workgroup barrier.
    __builtin_amdgcn_fence(__ATOMIC_ACQ_REL, "wavefront");

    // ---- P as A-fragment: row = n16, k(groups) = {g0..g0+7, g0+16..g0+23} ----
    AFrag pa;
    {
      const unsigned short* src = &Psh[w][n16][8 * half];
      pa.q4[0] = *(const uint4*)(src);
      pa.q4[1] = *(const uint4*)(src + 16);
    }

    // ---- out += P @ V : 4 d-tiles of 16, contraction over 32 keys ----
    #pragma unroll
    for (int dt = 0; dt < 4; ++dt) {
      AFrag bv;  // B: column n = d (16*dt+n16), k = key ({8*half+0..7, 16+8*half+0..7})
      const unsigned short* src = &VshT[buf][16 * dt + n16][8 * half];
      bv.q4[0] = *(const uint4*)(src);
      bv.q4[1] = *(const uint4*)(src + 16);
      acc[dt] = __builtin_amdgcn_wmma_f32_16x16x32_bf16(
          false, pa.v, false, bv.v, (short)0, acc[dt], false, false);
    }

    // ---- convert + store the prefetched next chunk into the other buffer ----
    if (t < 9) kv_store(Ksh[buf ^ 1], VshT[buf ^ 1], tid, pre);
    __syncthreads();   // orders: this iter's LDS reads < next iter's overwrite
  }

  // ---- epilogue: divide by softmax denominator, store f32 ----
  #pragma unroll
  for (int r = 0; r < 8; ++r) {
    const float  inv = __builtin_amdgcn_rcpf(lrow[r]);
    const size_t row = bhOff + (size_t)(qbase + 16 * w + 8 * half + r) * D_HEAD;
    #pragma unroll
    for (int dt = 0; dt < 4; ++dt)
      Og[row + 16 * dt + n16] = acc[dt][r] * inv;
  }
}

extern "C" void kernel_launch(void* const* d_in, const int* in_sizes, int n_in,
                              void* d_out, int out_size, void* d_ws, size_t ws_size,
                              hipStream_t stream) {
  const float* Q = (const float*)d_in[0];
  const float* K = (const float*)d_in[1];
  const float* V = (const float*)d_in[2];
  float* O = (float*)d_out;

  const int Bn = 4, Hn = 16;
  dim3 grid(S_LEN / 64, Bn * Hn);   // (64, 64)
  dim3 block(128);                  // 4 waves of 32
  swa_flash_wmma_kernel<<<grid, block, 0, stream>>>(Q, K, V, O);
}
